// GAT_22943715295510
// MI455X (gfx1250) — matrix operand order repaired and checked by reading, not compiled, not run on hardware
//
#include <hip/hip_runtime.h>

typedef __attribute__((ext_vector_type(16))) _Float16 v16h;
typedef __attribute__((ext_vector_type(8)))  _Float16 v8h;
typedef __attribute__((ext_vector_type(8)))  float    v8f;

// ---------------------------------------------------------------------------
// f32 -> f16 conversion
// ---------------------------------------------------------------------------
__global__ void k_f32_to_f16(const float* __restrict__ src,
                             _Float16* __restrict__ dst, int n) {
  int id = blockIdx.x * blockDim.x + threadIdx.x;
  if (id < n) dst[id] = (_Float16)src[id];
}

// ---------------------------------------------------------------------------
// WMMA GEMM: C[M,Nc] = A[M,K](f16) * B[K,Nc](f16) (+ bias[Nc]), f32 accumulate
// One wave per 16x64 output tile: 4 accumulators share one A fragment per
// 32-deep k-step -> 4 back-to-back independent v_wmma_f32_16x16x32_f16.
// M mult of 16, Nc mult of 64, K mult of 32. A,B row-major. Pure pointer-
// increment addressing; unconditional global_prefetch_b8 of the next B slab.
// ---------------------------------------------------------------------------
__global__ __launch_bounds__(128)
void k_wmma_gemm(const _Float16* __restrict__ A, const _Float16* __restrict__ B,
                 const float* __restrict__ bias, float* __restrict__ C,
                 int M, int Nc, int K) {
  const int tilesN = Nc >> 6;                       // 64-wide N tiles
  const int totalTiles = (M >> 4) * tilesN;
  const int wave = blockIdx.x * (blockDim.x >> 5) + (threadIdx.x >> 5);
  if (wave >= totalTiles) return;                   // wave-uniform: EXEC full
  const int m0   = (wave / tilesN) << 4;
  const int n0   = (wave % tilesN) << 6;
  const int lane = threadIdx.x & 31;
  const int hi   = lane >> 4;                       // 0: lanes 0-15, 1: 16-31
  const int l15  = lane & 15;

  // A: lane holds row m = m0+l15; K-subsets per ISA 16-bit A layout
  const _Float16* ap = A + (size_t)(m0 + l15) * K + hi * 8;
  // B: lane holds column n = n0+l15; K-subsets match the A layout
  const _Float16* bp = B + (size_t)(hi * 8) * Nc + n0 + l15;
  const size_t bstep = (size_t)32 * Nc;             // halfs per 32-deep k-step

  v8f acc0 = {}, acc1 = {}, acc2 = {}, acc3 = {};
  for (int k0 = 0; k0 < K; k0 += 32) {
    v8h a0 = *(const v8h*)ap;                       // K = base .. base+7
    v8h a1 = *(const v8h*)(ap + 16);                // K = base+16 .. base+23
    v16h af;
#pragma unroll
    for (int i = 0; i < 8; ++i) { af[i] = a0[i]; af[i + 8] = a1[i]; }

    v16h b0, b1, b2, b3;
#pragma unroll
    for (int i = 0; i < 16; ++i) {
      const _Float16* r = bp + (size_t)((i & 7) + (i >> 3) * 16) * Nc;
      b0[i] = r[0]; b1[i] = r[16]; b2[i] = r[32]; b3[i] = r[48];
    }
    __builtin_prefetch(bp + bstep, 0, 1);           // global_prefetch_b8

    acc0 = __builtin_amdgcn_wmma_f32_16x16x32_f16(false, af, false, b0, (short)0, acc0, false, false);
    acc1 = __builtin_amdgcn_wmma_f32_16x16x32_f16(false, af, false, b1, (short)0, acc1, false, false);
    acc2 = __builtin_amdgcn_wmma_f32_16x16x32_f16(false, af, false, b2, (short)0, acc2, false, false);
    acc3 = __builtin_amdgcn_wmma_f32_16x16x32_f16(false, af, false, b3, (short)0, acc3, false, false);

    ap += 32;                                       // +64 bytes
    bp += bstep;                                    // +32*Nc halfs
  }

  v8f accs[4] = {acc0, acc1, acc2, acc3};
#pragma unroll
  for (int nb = 0; nb < 4; ++nb) {
    int col = n0 + nb * 16 + l15;
    float bv = bias ? bias[col] : 0.0f;
#pragma unroll
    for (int j = 0; j < 8; ++j) {
      int row = m0 + hi * 8 + j;                    // C/D VGPR layout
      C[(size_t)row * Nc + col] = accs[nb][j] + bv;
    }
  }
}

// ---------------------------------------------------------------------------
// Attention dots: a_src[n,h] = <h[n,h,:], att_src[h,:]>, same for dst
// ---------------------------------------------------------------------------
__global__ void k_att_dot(const float* __restrict__ h,
                          const float* __restrict__ att_src,
                          const float* __restrict__ att_dst,
                          float* __restrict__ a_src, float* __restrict__ a_dst,
                          int n, int H, int C) {
  int id = blockIdx.x * blockDim.x + threadIdx.x;
  if (id >= n * H) return;
  int node = id / H, hd = id % H;
  const float* hp = h + (size_t)node * H * C + (size_t)hd * C;
  const float* as = att_src + (size_t)hd * C;
  const float* ad = att_dst + (size_t)hd * C;
  float s = 0.0f, d = 0.0f;
  for (int c = 0; c < C; ++c) { s += hp[c] * as[c]; d += hp[c] * ad[c]; }
  a_src[id] = s; a_dst[id] = d;
}

// ---------------------------------------------------------------------------
// Ordered-uint encoding so atomicMax(u32) == float max
// ---------------------------------------------------------------------------
__device__ __forceinline__ unsigned f2key(float f) {
  unsigned u = __float_as_uint(f);
  return (u & 0x80000000u) ? ~u : (u | 0x80000000u);
}
__device__ __forceinline__ float key2f(unsigned k) {
  return (k & 0x80000000u) ? __uint_as_float(k & 0x7fffffffu)
                           : __uint_as_float(~k);
}

__global__ void k_edge_logits(const int* __restrict__ src, const int* __restrict__ dst,
                              const float* __restrict__ a_src, const float* __restrict__ a_dst,
                              float* __restrict__ e, unsigned* __restrict__ mkey,
                              int E, int H) {
  int id = blockIdx.x * blockDim.x + threadIdx.x;
  if (id >= E * H) return;
  int eid = id / H, hd = id % H;
  int s = src[eid], d = dst[eid];
  float v = a_src[s * H + hd] + a_dst[d * H + hd];
  v = v > 0.0f ? v : 0.2f * v;                // leaky_relu(0.2)
  e[id] = v;
  atomicMax(&mkey[d * H + hd], f2key(v));
}

__global__ void k_m_finalize(const unsigned* __restrict__ mkey,
                             float* __restrict__ mval, int total) {
  int id = blockIdx.x * blockDim.x + threadIdx.x;
  if (id >= total) return;
  float f = key2f(mkey[id]);                  // key 0 => NaN => no-edge node
  unsigned b = __float_as_uint(f);
  bool finite = ((b >> 23) & 0xFFu) != 0xFFu;
  mval[id] = finite ? f : 0.0f;
}

__global__ void k_edge_exp(const int* __restrict__ dst, float* __restrict__ e,
                           const float* __restrict__ mval, float* __restrict__ denom,
                           int E, int H) {
  int id = blockIdx.x * blockDim.x + threadIdx.x;
  if (id >= E * H) return;
  int eid = id / H, hd = id % H;
  float ex = __expf(e[id] - mval[dst[eid] * H + hd]);
  e[id] = ex;
  atomicAdd(&denom[dst[eid] * H + hd], ex);
}

// out[dst, h, c] += h[src, h, c] * alpha(e, h)  (one thread per edge-feature)
__global__ void k_edge_scatter(const int* __restrict__ src, const int* __restrict__ dst,
                               const float* __restrict__ h, const float* __restrict__ ex,
                               const float* __restrict__ denom, float* __restrict__ out,
                               int E, int H, int C) {
  int id = blockIdx.x * blockDim.x + threadIdx.x;
  int HC = H * C;
  if (id >= E * HC) return;
  int eid = id / HC, f = id % HC, hd = f / C;
  int s = src[eid], d = dst[eid];
  float alpha = ex[eid * H + hd] / (denom[d * H + hd] + 1e-16f);
  atomicAdd(&out[(size_t)d * HC + f], h[(size_t)s * HC + f] * alpha);
}

// ---------------------------------------------------------------------------
// act = elu(gat + gat_bias + lin), write next-layer f16 activations
// ---------------------------------------------------------------------------
__global__ void k_combine_elu(const float* __restrict__ gat, const float* __restrict__ lin,
                              const float* __restrict__ bias, _Float16* __restrict__ actH,
                              int total, int F) {
  int id = blockIdx.x * blockDim.x + threadIdx.x;
  if (id >= total) return;
  float v = gat[id] + lin[id] + bias[id % F];
  v = v > 0.0f ? v : expm1f(v);
  actH[id] = (_Float16)v;
}

// out = mean_h(gat) + gat_bias + lin   (final layer, concat=False)
__global__ void k_combine_mean(const float* __restrict__ gat, const float* __restrict__ lin,
                               const float* __restrict__ bias, float* __restrict__ out,
                               int n, int H, int C) {
  int id = blockIdx.x * blockDim.x + threadIdx.x;
  if (id >= n * C) return;
  int node = id / C, c = id % C;
  float acc = 0.0f;
  for (int hd = 0; hd < H; ++hd) acc += gat[(size_t)node * H * C + hd * C + c];
  out[id] = acc * (1.0f / H) + lin[(size_t)node * C + c] + bias[c];
}

// ---------------------------------------------------------------------------
extern "C" void kernel_launch(void* const* d_in, const int* in_sizes, int n_in,
                              void* d_out, int out_size, void* d_ws, size_t ws_size,
                              hipStream_t stream) {
  (void)n_in; (void)out_size; (void)ws_size;
  const float* x  = (const float*)d_in[0];
  const int*   ei = (const int*)d_in[1];
  const float* W[3]  = {(const float*)d_in[2],  (const float*)d_in[8],  (const float*)d_in[14]};
  const float* AS[3] = {(const float*)d_in[3],  (const float*)d_in[9],  (const float*)d_in[15]};
  const float* AD[3] = {(const float*)d_in[4],  (const float*)d_in[10], (const float*)d_in[16]};
  const float* GB[3] = {(const float*)d_in[5],  (const float*)d_in[11], (const float*)d_in[17]};
  const float* LW[3] = {(const float*)d_in[6],  (const float*)d_in[12], (const float*)d_in[18]};
  const float* LB[3] = {(const float*)d_in[7],  (const float*)d_in[13], (const float*)d_in[19]};

  const int N = in_sizes[0] / 256;   // 50000 (multiple of 16)
  const int E = in_sizes[1] / 2;     // 300000
  const int* src = ei;
  const int* dst = ei + E;

  // ---- workspace carve (256B aligned) ----
  char* wp = (char*)d_ws;
  auto carve = [&](size_t bytes) -> void* {
    void* r = (void*)wp;
    wp += (bytes + 255) & ~(size_t)255;
    return r;
  };
  _Float16* actH  = (_Float16*)carve((size_t)N * 256 * sizeof(_Float16));
  _Float16* wH    = (_Float16*)carve((size_t)256 * 512 * sizeof(_Float16));
  float*    hBuf  = (float*)carve((size_t)N * 512 * sizeof(float));
  float*    linB  = (float*)carve((size_t)N * 256 * sizeof(float));
  float*    gatB  = (float*)carve((size_t)N * 512 * sizeof(float));
  float*    aSrc  = (float*)carve((size_t)N * 4 * sizeof(float));
  float*    aDst  = (float*)carve((size_t)N * 4 * sizeof(float));
  unsigned* mKey  = (unsigned*)carve((size_t)N * 4 * sizeof(unsigned));
  float*    mVal  = (float*)carve((size_t)N * 4 * sizeof(float));
  float*    denom = (float*)carve((size_t)N * 4 * sizeof(float));
  float*    eBuf  = (float*)carve((size_t)E * 4 * sizeof(float));

  auto nb = [](long n, int t) { return (unsigned)((n + t - 1) / t); };

  // layer 0 input activations -> f16
  k_f32_to_f16<<<nb((long)N * 256, 256), 256, 0, stream>>>(x, actH, N * 256);

  for (int l = 0; l < 3; ++l) {
    const int H  = 4;
    const int C  = (l == 2) ? 128 : 64;
    const int Fin = 256;
    const int FG = H * C;                  // 256, 256, 512
    const int FL = (l == 2) ? 128 : 256;   // linear skip width

    // h = act @ W   (WMMA, f32 accumulate)
    k_f32_to_f16<<<nb((long)Fin * FG, 256), 256, 0, stream>>>(W[l], wH, Fin * FG);
    {
      int tiles = (N >> 4) * (FG >> 6);
      k_wmma_gemm<<<nb(tiles, 4), 128, 0, stream>>>(actH, wH, nullptr, hBuf, N, FG, Fin);
    }
    // lin = act @ LW + lb   (WMMA, bias fused)
    k_f32_to_f16<<<nb((long)Fin * FL, 256), 256, 0, stream>>>(LW[l], wH, Fin * FL);
    {
      int tiles = (N >> 4) * (FL >> 6);
      k_wmma_gemm<<<nb(tiles, 4), 128, 0, stream>>>(actH, wH, LB[l], linB, N, FL, Fin);
    }

    // attention logits per node
    k_att_dot<<<nb((long)N * H, 256), 256, 0, stream>>>(hBuf, AS[l], AD[l], aSrc, aDst, N, H, C);

    // segment softmax over edges (max -> exp -> sum)
    hipMemsetAsync(mKey, 0, (size_t)N * H * sizeof(unsigned), stream);
    k_edge_logits<<<nb((long)E * H, 256), 256, 0, stream>>>(src, dst, aSrc, aDst, eBuf, mKey, E, H);
    k_m_finalize<<<nb((long)N * H, 256), 256, 0, stream>>>(mKey, mVal, N * H);
    hipMemsetAsync(denom, 0, (size_t)N * H * sizeof(float), stream);
    k_edge_exp<<<nb((long)E * H, 256), 256, 0, stream>>>(dst, eBuf, mVal, denom, E, H);

    // message scatter
    hipMemsetAsync(gatB, 0, (size_t)N * FG * sizeof(float), stream);
    k_edge_scatter<<<nb((long)E * FG, 256), 256, 0, stream>>>(src, dst, hBuf, eBuf, denom, gatB, E, H, C);

    if (l < 2) {
      k_combine_elu<<<nb((long)N * FG, 256), 256, 0, stream>>>(gatB, linB, GB[l], actH, N * FG, FG);
    } else {
      k_combine_mean<<<nb((long)N * C, 256), 256, 0, stream>>>(gatB, linB, GB[l], (float*)d_out, N, H, C);
    }
  }
}
